// flaxGraphNetwork_24627342475954
// MI455X (gfx1250) — matrix operand order repaired and checked by reading, not compiled, not run on hardware
//
#include <hip/hip_runtime.h>
#include <cstdint>

// ---------------------------------------------------------------------------
// Sizes (compile-time constants matching the reference)
// ---------------------------------------------------------------------------
constexpr int kNodes = 100000;
constexpr int kEdges = 1600000;
constexpr int kL = 64;

// ---------------------------------------------------------------------------
// CDNA5 WMMA types
// ---------------------------------------------------------------------------
typedef __attribute__((ext_vector_type(16))) __bf16 v16bf;
typedef __attribute__((ext_vector_type(8)))  float  v8f;

static __device__ inline v8f wmma_bf16(v16bf a, v16bf b, v8f c) {
  // D = A(16x32 bf16) x B(32x16 bf16) + C(16x16 f32)
  return __builtin_amdgcn_wmma_f32_16x16x32_bf16(
      /*neg_a=*/false, a, /*neg_b=*/false, b,
      /*c_mod=*/(short)0, c, /*reuse_a=*/false, /*reuse_b=*/false);
}

// ---------------------------------------------------------------------------
// Scalar helpers
// ---------------------------------------------------------------------------
static __device__ inline unsigned short f2bf(float f) {
  unsigned u = __float_as_uint(f);
  unsigned r = (u + 0x7FFFu + ((u >> 16) & 1u)) >> 16;  // RNE
  return (unsigned short)r;
}

static __device__ inline float asinh_dev(float x) {
  float ax = fabsf(x);
  float r = logf(ax + sqrtf(fmaf(ax, ax, 1.0f)));
  return copysignf(r, x);
}

static __device__ inline float gelu_dev(float x) {  // jax.nn.gelu (tanh approx)
  float t = tanhf(0.7978845608028654f * (x + 0.044715f * x * x * x));
  return 0.5f * x * (1.0f + t);
}

// monotonic float<->uint encoding so float-max can use atomicMax(u32)
static __device__ inline unsigned fkey(float f) {
  unsigned u = __float_as_uint(f);
  return (u & 0x80000000u) ? ~u : (u | 0x80000000u);
}
static __device__ inline float fdekey(unsigned k) {
  unsigned u = (k & 0x80000000u) ? (k & 0x7FFFFFFFu) : ~k;
  return __uint_as_float(u);
}

// ---------------------------------------------------------------------------
// Fragment loaders
// A (16x32 bf16) from an LDS row of bf16 (stored as ushort), row-major in K.
//   lane l: rows m=l&15; slots 0..7 = K kbase+hk..+7, slots 8..15 = +16..+23
//   where hk = (l>>4)*8.  Two contiguous 16B chunks -> ds_load_b128 pairs.
// ---------------------------------------------------------------------------
static __device__ inline v16bf load_afrag(const unsigned short* row, int kbase, int hi) {
  union { uint4 q[2]; v16bf v; } f;
  const unsigned short* p = row + kbase + hi * 8;
  f.q[0] = *(const uint4*)(p);
  f.q[1] = *(const uint4*)(p + 16);
  return f.v;
}

// B (32x16 bf16) from a row-major [64][64] bf16 weight matrix in global mem.
//   element (k,n): lane = n + 16*((k/8)%2), slot = (k%8) + 8*(k/16)
static __device__ inline v16bf load_bfrag(const unsigned short* __restrict__ W,
                                          int kbase, int nbase, int lane) {
  union { unsigned short u[16]; v16bf v; } f;
  const unsigned n = (unsigned)(nbase + (lane & 15));
  const unsigned kb = (unsigned)(kbase + ((lane >> 4) << 3));
#pragma unroll
  for (unsigned s = 0; s < 16; ++s) f.u[s] = W[(kb + s + (s & 8u)) * 64u + n];
  return f.v;
}

// ---------------------------------------------------------------------------
// Kernel 0: zero accumulation buffers (must run every launch)
// ---------------------------------------------------------------------------
__global__ void zero_ws_kernel(float* __restrict__ sent, float* __restrict__ statsF,
                               unsigned* __restrict__ maxu, int sentN) {
  const int id = blockIdx.x * blockDim.x + threadIdx.x;
  const int stride = gridDim.x * blockDim.x;
  for (int i = id; i < sentN; i += stride) sent[i] = 0.0f;
  if (id < 256) statsF[id] = 0.0f;
  if (id < 128) maxu[id] = 0u;  // < fkey(any finite value)
}

// ---------------------------------------------------------------------------
// Kernel 1: weight prep — fold duplicated sender/receiver blocks, fold the
// constant global block into biases, convert GEMM weights to bf16.
// ---------------------------------------------------------------------------
__global__ void prep_kernel(const float* __restrict__ p_glob, const float* __restrict__ w_eg,
                            const float* __restrict__ b_eg,
                            const float* __restrict__ ew1, const float* __restrict__ eb1,
                            const float* __restrict__ ew2,
                            const float* __restrict__ nw1, const float* __restrict__ nb1,
                            const float* __restrict__ nw2,
                            unsigned short* __restrict__ W1e, unsigned short* __restrict__ W1n,
                            unsigned short* __restrict__ W2e,
                            unsigned short* __restrict__ Wn1, unsigned short* __restrict__ Wns,
                            unsigned short* __restrict__ W2n,
                            float* __restrict__ glob_emb, float* __restrict__ const_e,
                            float* __restrict__ const_n) {
  const int id = blockIdx.x * blockDim.x + threadIdx.x;
  if (id < 4096) {
    const int k = id >> 6, n = id & 63;
    W1e[id] = f2bf(ew1[k * 64 + n]);
    W1n[id] = f2bf(ew1[(64 + k) * 64 + n] + ew1[(128 + k) * 64 + n]);
    W2e[id] = f2bf(ew2[id]);
    Wn1[id] = f2bf(nw1[k * 64 + n]);
    Wns[id] = f2bf(nw1[(64 + k) * 64 + n] + nw1[(128 + k) * 64 + n]);
    W2n[id] = f2bf(nw2[id]);
  }
  if (id < 64) {
    const float pa = p_glob[0], pb = p_glob[1], pc = p_glob[2], pd = p_glob[3];
    const float g0 = asinh_dev((float)kNodes);
    const float g1 = asinh_dev((float)kEdges);
    const float a0 = pa * asinh_dev(pb * g0 + pc) + pd;
    const float a1 = pa * asinh_dev(pb * g1 + pc) + pd;
    glob_emb[id] = a0 * w_eg[id] + a1 * w_eg[64 + id] + b_eg[id];
    float ce = eb1[id], cn = nb1[id];
    for (int j = 0; j < 64; ++j) {
      const float gj = a0 * w_eg[j] + a1 * w_eg[64 + j] + b_eg[j];
      ce += gj * ew1[(192 + j) * 64 + id];
      cn += gj * nw1[(192 + j) * 64 + id];
    }
    const_e[id] = ce;
    const_n[id] = cn;
  }
}

// ---------------------------------------------------------------------------
// Kernel 2: node embedding (7 -> 64), cols 0..3 zeroed per reference.
// Writes f32 (node-MLP input) and bf16 (edge-kernel gather table; fits in L2).
// ---------------------------------------------------------------------------
__global__ void embed_kernel(const float* __restrict__ nodes, const float* __restrict__ p_node,
                             const float* __restrict__ w_en, const float* __restrict__ b_en,
                             float* __restrict__ embF, unsigned short* __restrict__ embB) {
  __shared__ float sw[512];  // [0:448) w_en (7x64 row-major), [448:512) b_en
  for (int i = threadIdx.x; i < 512; i += blockDim.x)
    sw[i] = (i < 448) ? w_en[i] : b_en[i - 448];
  __syncthreads();
  const int id = blockIdx.x * blockDim.x + threadIdx.x;
  if (id >= kNodes) return;
  const float pa = p_node[0], pb = p_node[1], pc = p_node[2], pd = p_node[3];
  const float t0 = pa * asinh_dev(pc) + pd;  // asinh_layer(0) for zeroed cols 0..3
  const unsigned ib = (unsigned)id * 7u;
  const float x4 = nodes[ib + 4];
  const float x5 = nodes[ib + 5];
  const float x6 = nodes[ib + 6];
  const float t4 = pa * asinh_dev(pb * x4 + pc) + pd;
  const float t5 = pa * asinh_dev(pb * x5 + pc) + pd;
  const float t6 = pa * asinh_dev(pb * x6 + pc) + pd;
  const unsigned ob = (unsigned)id * 64u;
  for (int j = 0; j < 64; ++j) {
    float v = sw[448 + j] + t0 * (sw[j] + sw[64 + j] + sw[128 + j] + sw[192 + j]) +
              t4 * sw[256 + j] + t5 * sw[320 + j] + t6 * sw[384 + j];
    embF[ob + j] = v;
    embB[ob + j] = f2bf(v);
  }
}

// ---------------------------------------------------------------------------
// Kernel 3: fused edge pipeline. One wave per 16-edge tile (grid-stride):
//   edge embed (3->64) + gather node_emb[sender] -> LDS bf16 (K=128)
//   -> WMMA layer1 (K=128, +const bias, gelu) -> WMMA layer2 (K=64, +eb2)
//   -> atomic scatter-add into sent[sender] + running sum/sumsq/max stats.
// All global hot-loop accesses use 32-bit offsets so the compiler can emit
// GVS addressing (SGPR base + 32-bit voffset) instead of 64-bit VGPR math.
// ---------------------------------------------------------------------------
__global__ __launch_bounds__(32) void edge_kernel(
    const float* __restrict__ edges, const int* __restrict__ senders,
    const float* __restrict__ p_edge, const float* __restrict__ w_ee,
    const float* __restrict__ b_ee, const unsigned short* __restrict__ nodes_bf,
    const unsigned short* __restrict__ W1e, const unsigned short* __restrict__ W1n,
    const unsigned short* __restrict__ W2, const float* __restrict__ cbias,
    const float* __restrict__ bias2, float* __restrict__ sent,
    float* __restrict__ st_sum, float* __restrict__ st_sq, unsigned* __restrict__ st_max) {
  const int lane = threadIdx.x;
  const int m = lane & 15, hi = lane >> 4, n0 = lane & 15;

  __shared__ __align__(16) unsigned short sAct[16][136];  // 128 used (+pad: 272B rows)
  __shared__ __align__(16) unsigned short sH[16][72];     // 64 used (+pad: 144B rows)
  __shared__ int sSnd[16];
  __shared__ float sC[384];  // [0:192) w_ee, [192:256) b_ee, [256:320) const_e, [320:384) eb2

  for (int i = lane; i < 384; i += 32) {
    float v;
    if (i < 192) v = w_ee[i];
    else if (i < 256) v = b_ee[i - 192];
    else if (i < 320) v = cbias[i - 256];
    else v = bias2[i - 320];
    sC[i] = v;
  }
  __syncthreads();

  // Register-resident weight fragments (loaded once per wave).
  v16bf w1[4][4];
#pragma unroll
  for (int ks = 0; ks < 4; ++ks)
#pragma unroll
    for (int nt = 0; nt < 4; ++nt)
      w1[ks][nt] = load_bfrag(ks < 2 ? W1e : W1n, (ks & 1) * 32, nt * 16, lane);
  v16bf w2[2][4];
#pragma unroll
  for (int ks = 0; ks < 2; ++ks)
#pragma unroll
    for (int nt = 0; nt < 4; ++nt) w2[ks][nt] = load_bfrag(W2, ks * 32, nt * 16, lane);

  const float pa = p_edge[0], pb = p_edge[1], pc = p_edge[2], pd = p_edge[3];
  const v8f zero8 = {0.f, 0.f, 0.f, 0.f, 0.f, 0.f, 0.f, 0.f};

  float esum[4] = {0.f, 0.f, 0.f, 0.f};
  float esq[4] = {0.f, 0.f, 0.f, 0.f};
  float emx[4] = {-3.0e38f, -3.0e38f, -3.0e38f, -3.0e38f};

  const int ntiles = kEdges / 16;
  for (int tile = blockIdx.x; tile < ntiles; tile += gridDim.x) {
    const int tb = tile * 16;
    if (lane < 16) sSnd[lane] = senders[tb + lane];
    __syncthreads();

    // Gather node embedding row (bf16, L2-resident table) -> act cols 64..127.
    {
      const unsigned gofs = (unsigned)sSnd[m] * 64u + (unsigned)(hi * 32);
      const uint4* src = (const uint4*)(nodes_bf + gofs);
      uint4* dst = (uint4*)(&sAct[m][64 + hi * 32]);
      dst[0] = src[0]; dst[1] = src[1]; dst[2] = src[2]; dst[3] = src[3];
    }
    // Edge embedding (3->64); this lane produces cols [hi*32, hi*32+32).
    {
      const unsigned eofs = (unsigned)(tb + m) * 3u;
      const float x0 = edges[eofs + 0];
      const float x1 = edges[eofs + 1];
      const float x2 = edges[eofs + 2];
      const float t0 = pa * asinh_dev(pb * x0 + pc) + pd;
      const float t1 = pa * asinh_dev(pb * x1 + pc) + pd;
      const float t2 = pa * asinh_dev(pb * x2 + pc) + pd;
#pragma unroll
      for (int jj = 0; jj < 16; ++jj) {
        const int j = hi * 32 + jj * 2;
        const float v0 = sC[192 + j] + t0 * sC[j] + t1 * sC[64 + j] + t2 * sC[128 + j];
        const float v1 = sC[193 + j] + t0 * sC[j + 1] + t1 * sC[65 + j] + t2 * sC[129 + j];
        *(unsigned*)(&sAct[m][j]) = (unsigned)f2bf(v0) | ((unsigned)f2bf(v1) << 16);
      }
    }
    __syncthreads();

    // Layer 1: (16x128) @ (128x64)
    v8f acc[4] = {zero8, zero8, zero8, zero8};
#pragma unroll
    for (int ks = 0; ks < 4; ++ks) {
      const v16bf a = load_afrag(&sAct[m][0], ks * 32, hi);
#pragma unroll
      for (int nt = 0; nt < 4; ++nt) acc[nt] = wmma_bf16(a, w1[ks][nt], acc[nt]);
    }
    // bias (folded glob contribution) + gelu -> sH as bf16
#pragma unroll
    for (int nt = 0; nt < 4; ++nt) {
      const float b = sC[256 + nt * 16 + n0];
#pragma unroll
      for (int r = 0; r < 8; ++r)
        sH[r + 8 * hi][nt * 16 + n0] = f2bf(gelu_dev(acc[nt][r] + b));
    }
    __syncthreads();

    // Layer 2: (16x64) @ (64x64)
    v8f out[4] = {zero8, zero8, zero8, zero8};
#pragma unroll
    for (int ks = 0; ks < 2; ++ks) {
      const v16bf a = load_afrag(&sH[m][0], ks * 32, hi);
#pragma unroll
      for (int nt = 0; nt < 4; ++nt) out[nt] = wmma_bf16(a, w2[ks][nt], out[nt]);
    }

    // Epilogue: segment-sum scatter (saddr + 32-bit voffset atomics) + stats.
    unsigned rowOfs[8];
#pragma unroll
    for (int r = 0; r < 8; ++r) rowOfs[r] = (unsigned)sSnd[r + 8 * hi] * 64u;

#pragma unroll
    for (int r = 0; r < 8; ++r) {
#pragma unroll
      for (int nt = 0; nt < 4; ++nt) {
        const unsigned n = (unsigned)(nt * 16 + n0);
        const float v = out[nt][r] + sC[320 + n];
        atomicAdd(&sent[rowOfs[r] + n], v);
        esum[nt] += v;
        esq[nt] += v * v;
        emx[nt] = fmaxf(emx[nt], v);
      }
    }
    __syncthreads();
  }

  // Fold lane pairs (lane, lane^16 own the same column) and flush once.
#pragma unroll
  for (int nt = 0; nt < 4; ++nt) {
    const float s = esum[nt] + __shfl_xor(esum[nt], 16, 32);
    const float q = esq[nt] + __shfl_xor(esq[nt], 16, 32);
    const float x = fmaxf(emx[nt], __shfl_xor(emx[nt], 16, 32));
    if (lane < 16) {
      const int n = nt * 16 + lane;
      atomicAdd(&st_sum[n], s);
      atomicAdd(&st_sq[n], q);
      atomicMax(&st_max[n], fkey(x));
    }
  }
}

// ---------------------------------------------------------------------------
// Kernel 4: fused node pipeline. Same WMMA structure, inputs are the node
// embedding (f32) and the aggregated `sent` (f32); no scatter, stats only.
// ---------------------------------------------------------------------------
__global__ __launch_bounds__(32) void node_kernel(
    const float* __restrict__ embF, const float* __restrict__ sentF,
    const unsigned short* __restrict__ W1a, const unsigned short* __restrict__ W1b,
    const unsigned short* __restrict__ W2, const float* __restrict__ cbias,
    const float* __restrict__ bias2, float* __restrict__ st_sum,
    float* __restrict__ st_sq, unsigned* __restrict__ st_max) {
  const int lane = threadIdx.x;
  const int m = lane & 15, hi = lane >> 4, n0 = lane & 15;

  __shared__ __align__(16) unsigned short sAct[16][136];
  __shared__ __align__(16) unsigned short sH[16][72];
  __shared__ float sC[128];  // [0:64) const_n, [64:128) nb2
  for (int i = lane; i < 128; i += 32) sC[i] = (i < 64) ? cbias[i] : bias2[i - 64];
  __syncthreads();

  v16bf w1[4][4];
#pragma unroll
  for (int ks = 0; ks < 4; ++ks)
#pragma unroll
    for (int nt = 0; nt < 4; ++nt)
      w1[ks][nt] = load_bfrag(ks < 2 ? W1a : W1b, (ks & 1) * 32, nt * 16, lane);
  v16bf w2[2][4];
#pragma unroll
  for (int ks = 0; ks < 2; ++ks)
#pragma unroll
    for (int nt = 0; nt < 4; ++nt) w2[ks][nt] = load_bfrag(W2, ks * 32, nt * 16, lane);

  const v8f zero8 = {0.f, 0.f, 0.f, 0.f, 0.f, 0.f, 0.f, 0.f};
  float ssum[4] = {0.f, 0.f, 0.f, 0.f};
  float ssq[4] = {0.f, 0.f, 0.f, 0.f};
  float smx[4] = {-3.0e38f, -3.0e38f, -3.0e38f, -3.0e38f};

  const int ntiles = kNodes / 16;
  for (int tile = blockIdx.x; tile < ntiles; tile += gridDim.x) {
    const int node = tile * 16 + m;
    // hi==0 stages node_emb -> cols 0..63; hi==1 stages sent -> cols 64..127
    const float* srcBase = (hi == 0) ? embF : sentF;
    const unsigned srcOfs = (unsigned)node * 64u;
#pragma unroll
    for (int j = 0; j < 64; j += 2) {
      const unsigned pk =
          (unsigned)f2bf(srcBase[srcOfs + j]) | ((unsigned)f2bf(srcBase[srcOfs + j + 1]) << 16);
      *(unsigned*)(&sAct[m][hi * 64 + j]) = pk;
    }
    __syncthreads();

    v8f acc[4] = {zero8, zero8, zero8, zero8};
#pragma unroll
    for (int ks = 0; ks < 4; ++ks) {
      const v16bf a = load_afrag(&sAct[m][0], ks * 32, hi);
#pragma unroll
      for (int nt = 0; nt < 4; ++nt) acc[nt] = wmma_bf16(a, w1[ks][nt], acc[nt]);
    }
#pragma unroll
    for (int nt = 0; nt < 4; ++nt) {
      const float b = sC[nt * 16 + n0];
#pragma unroll
      for (int r = 0; r < 8; ++r)
        sH[r + 8 * hi][nt * 16 + n0] = f2bf(gelu_dev(acc[nt][r] + b));
    }
    __syncthreads();

    v8f out[4] = {zero8, zero8, zero8, zero8};
#pragma unroll
    for (int ks = 0; ks < 2; ++ks) {
      const v16bf a = load_afrag(&sH[m][0], ks * 32, hi);
#pragma unroll
      for (int nt = 0; nt < 4; ++nt) out[nt] = wmma_bf16(a, w2[ks][nt], out[nt]);
    }
#pragma unroll
    for (int nt = 0; nt < 4; ++nt) {
      const float b2 = sC[64 + nt * 16 + n0];
#pragma unroll
      for (int r = 0; r < 8; ++r) {
        const float v = out[nt][r] + b2;
        ssum[nt] += v;
        ssq[nt] += v * v;
        smx[nt] = fmaxf(smx[nt], v);
      }
    }
    __syncthreads();
  }

#pragma unroll
  for (int nt = 0; nt < 4; ++nt) {
    const float s = ssum[nt] + __shfl_xor(ssum[nt], 16, 32);
    const float q = ssq[nt] + __shfl_xor(ssq[nt], 16, 32);
    const float x = fmaxf(smx[nt], __shfl_xor(smx[nt], 16, 32));
    if (lane < 16) {
      const int n = nt * 16 + lane;
      atomicAdd(&st_sum[n], s);
      atomicAdd(&st_sq[n], q);
      atomicMax(&st_max[n], fkey(x));
    }
  }
}

// ---------------------------------------------------------------------------
// Kernel 5: global MLP on the 1x576 aggregate vector (tiny; one workgroup).
// ---------------------------------------------------------------------------
__global__ void finalize_kernel(const float* __restrict__ glob_emb,
                                const float* __restrict__ n_sum, const float* __restrict__ n_sq,
                                const unsigned* __restrict__ n_max,
                                const float* __restrict__ e_sum, const float* __restrict__ e_sq,
                                const unsigned* __restrict__ e_max,
                                const float* __restrict__ gw1, const float* __restrict__ gb1,
                                const float* __restrict__ gw2, const float* __restrict__ gb2,
                                const float* __restrict__ gw3, const float* __restrict__ gb3,
                                float* __restrict__ out) {
  __shared__ float gin[576];
  __shared__ float h1[64];
  __shared__ float h2[64];
  const int t = threadIdx.x;  // 64 threads
  {
    const float invN = 1.0f / (float)kNodes;
    const float invE = 1.0f / (float)kEdges;
    float s = n_sum[t], mn = s * invN;
    float var = n_sq[t] * invN - mn * mn;
    gin[t] = asinh_dev(s);
    gin[64 + t] = asinh_dev(mn);
    gin[128 + t] = asinh_dev(var);
    gin[192 + t] = asinh_dev(fdekey(n_max[t]));
    s = e_sum[t]; mn = s * invE;
    var = e_sq[t] * invE - mn * mn;
    gin[256 + t] = asinh_dev(s);
    gin[320 + t] = asinh_dev(mn);
    gin[384 + t] = asinh_dev(var);
    gin[448 + t] = asinh_dev(fdekey(e_max[t]));
    gin[512 + t] = glob_emb[t];
  }
  __syncthreads();
  float a = gb1[t];
  for (int i = 0; i < 576; ++i) a += gin[i] * gw1[i * 64 + t];
  h1[t] = gelu_dev(a);
  __syncthreads();
  a = gb2[t];
  for (int i = 0; i < 64; ++i) a += h1[i] * gw2[i * 64 + t];
  h2[t] = gelu_dev(a);
  __syncthreads();
  if (t < 2) {
    float r = gb3[t];
    for (int i = 0; i < 64; ++i) r += h2[i] * gw3[i * 2 + t];
    out[t] = r;
  }
}

// ---------------------------------------------------------------------------
// Host launcher
// ---------------------------------------------------------------------------
extern "C" void kernel_launch(void* const* d_in, const int* in_sizes, int n_in,
                              void* d_out, int out_size, void* d_ws, size_t ws_size,
                              hipStream_t stream) {
  (void)in_sizes; (void)n_in; (void)out_size; (void)ws_size;
  const float* nodes  = (const float*)d_in[0];
  const float* edges  = (const float*)d_in[1];
  const int*   snd    = (const int*)d_in[2];
  const float* p_node = (const float*)d_in[3];
  const float* w_en   = (const float*)d_in[4];
  const float* b_en   = (const float*)d_in[5];
  const float* p_edge = (const float*)d_in[6];
  const float* w_ee   = (const float*)d_in[7];
  const float* b_ee   = (const float*)d_in[8];
  const float* p_glob = (const float*)d_in[9];
  const float* w_eg   = (const float*)d_in[10];
  const float* b_eg   = (const float*)d_in[11];
  const float* ew1    = (const float*)d_in[12];
  const float* eb1    = (const float*)d_in[13];
  const float* ew2    = (const float*)d_in[14];
  const float* eb2    = (const float*)d_in[15];
  const float* nw1    = (const float*)d_in[16];
  const float* nb1    = (const float*)d_in[17];
  const float* nw2    = (const float*)d_in[18];
  const float* nb2    = (const float*)d_in[19];
  const float* gw1    = (const float*)d_in[20];
  const float* gb1    = (const float*)d_in[21];
  const float* gw2    = (const float*)d_in[22];
  const float* gb2    = (const float*)d_in[23];
  const float* gw3    = (const float*)d_in[24];
  const float* gb3    = (const float*)d_in[25];
  float* out = (float*)d_out;

  char* ws = (char*)d_ws;
  size_t off = 0;
  auto alloc = [&](size_t bytes) -> void* {
    void* p = (void*)(ws + off);
    off += (bytes + 255) & ~(size_t)255;
    return p;
  };
  float* embF = (float*)alloc(sizeof(float) * (size_t)kNodes * kL);
  unsigned short* embB = (unsigned short*)alloc(sizeof(unsigned short) * (size_t)kNodes * kL);
  float* sent = (float*)alloc(sizeof(float) * (size_t)kNodes * kL);
  unsigned short* W1e = (unsigned short*)alloc(8192);
  unsigned short* W1n = (unsigned short*)alloc(8192);
  unsigned short* W2e = (unsigned short*)alloc(8192);
  unsigned short* Wn1 = (unsigned short*)alloc(8192);
  unsigned short* Wns = (unsigned short*)alloc(8192);
  unsigned short* W2n = (unsigned short*)alloc(8192);
  float* glob_emb = (float*)alloc(sizeof(float) * 64);
  float* const_e = (float*)alloc(sizeof(float) * 64);
  float* const_n = (float*)alloc(sizeof(float) * 64);
  float* statsF = (float*)alloc(sizeof(float) * 256);   // e_sum, e_sq, n_sum, n_sq
  unsigned* maxu = (unsigned*)alloc(sizeof(unsigned) * 128);  // e_max, n_max

  float* e_sum = statsF;
  float* e_sq = statsF + 64;
  float* n_sum = statsF + 128;
  float* n_sq = statsF + 192;
  unsigned* e_max = maxu;
  unsigned* n_max = maxu + 64;

  zero_ws_kernel<<<25000, 256, 0, stream>>>(sent, statsF, maxu, kNodes * kL);
  prep_kernel<<<16, 256, 0, stream>>>(p_glob, w_eg, b_eg, ew1, eb1, ew2, nw1, nb1, nw2,
                                      W1e, W1n, W2e, Wn1, Wns, W2n,
                                      glob_emb, const_e, const_n);
  embed_kernel<<<(kNodes + 255) / 256, 256, 0, stream>>>(nodes, p_node, w_en, b_en, embF, embB);
  edge_kernel<<<4096, 32, 0, stream>>>(edges, snd, p_edge, w_ee, b_ee, embB,
                                       W1e, W1n, W2e, const_e, eb2, sent,
                                       e_sum, e_sq, e_max);
  node_kernel<<<2048, 32, 0, stream>>>(embF, sent, Wn1, Wns, W2n, const_n, nb2,
                                       n_sum, n_sq, n_max);
  finalize_kernel<<<1, 64, 0, stream>>>(glob_emb, n_sum, n_sq, n_max, e_sum, e_sq, e_max,
                                        gw1, gb1, gw2, gb2, gw3, gb3, out);
}
